// FullyConnectedThermalGNN_8358006358348
// MI455X (gfx1250) — matrix operand order repaired and checked by reading, not compiled, not run on hardware
//
#include <hip/hip_runtime.h>

typedef __attribute__((ext_vector_type(16))) _Float16 v16h;
typedef __attribute__((ext_vector_type(8)))  _Float16 v8h;
typedef __attribute__((ext_vector_type(8)))  float    v8f;

#define NN      10000      // nodes
#define NE      400000     // edges
#define HDIM    128
#define NIN     6
#define EDIM    3
#define NL      4
#define LN_EPS  1e-5f

// ---------------------------------------------------------------------------
// A-fragment loader: 16x32 f16 tile from LDS (row-major, rowStride halves).
// Layout: lanes 0-15 -> M=lane, elems 0-15:K=kt*32+0..15
//         lanes 16-31 -> M=lane-16, elems 0-15:K=kt*32+16..31
// ---------------------------------------------------------------------------
__device__ __forceinline__ v16h load_a_frag(const _Float16* sA, int rowStride,
                                            int kt, int lane) {
  int m = lane & 15;
  int kbase = kt * 32 + ((lane & 16) ? 16 : 0);
  const _Float16* p = sA + m * rowStride + kbase;
  v8h lo = *(const v8h*)(p);
  v8h hi = *(const v8h*)(p + 8);
  return __builtin_shufflevector(lo, hi, 0,1,2,3,4,5,6,7,8,9,10,11,12,13,14,15);
}

// ---------------------------------------------------------------------------
// Pack fp32 row-major weight W[K][Ncols] into f16 B-fragments.
// One wave per 32x16 fragment; frag = kt*(Ncols/16)+nt; lane l holds 16
// contiguous halves: lanes 0-15 K=kt*32+i, lanes 16-31 K=kt*32+16+i, N=l&15.
// Zero-fills rows >= K (padding for msg_w1's K=259 -> 288).
// ---------------------------------------------------------------------------
__global__ void gnn_pack_b_kernel(const float* __restrict__ W,
                                  _Float16* __restrict__ out,
                                  int K, int Ncols) {
  int frag = blockIdx.x;
  int lane = threadIdx.x;
  int ntiles = Ncols >> 4;
  int kt = frag / ntiles, nt = frag - kt * ntiles;
  int n  = lane & 15;
  int kb = kt * 32 + ((lane & 16) ? 16 : 0);
  v16h pack;
#pragma unroll
  for (int i = 0; i < 16; ++i) {
    int wk = kb + i;
    float v = (wk < K) ? W[wk * Ncols + nt * 16 + n] : 0.f;
    pack[i] = (_Float16)v;
  }
  *(v16h*)(out + frag * 512 + lane * 16) = pack;
}

// ---------------------------------------------------------------------------
// Utility kernels
// ---------------------------------------------------------------------------
__global__ void gnn_zero_kernel(float* __restrict__ p, int n) {
  int i = blockIdx.x * blockDim.x + threadIdx.x;
  if (i < n) p[i] = 0.f;
}

__global__ void gnn_count_kernel(const int* __restrict__ tgt,
                                 float* __restrict__ counts) {
  int e = blockIdx.x * blockDim.x + threadIdx.x;
  if (e < NE) atomicAdd(&counts[tgt[e]], 1.f);
}

__global__ void gnn_inv_kernel(const float* __restrict__ counts,
                               float* __restrict__ inv) {
  int n = blockIdx.x * blockDim.x + threadIdx.x;
  if (n < NN) inv[n] = 1.f / fmaxf(counts[n], 1.f);
}

// Encoder first layer: relu(x @ w1 + b1), K=6 so scalar FMA is fine.
// Emits f16 directly (consumed only as a WMMA A operand).
__global__ void gnn_encoder1_kernel(const float* __restrict__ x,
                                    const float* __restrict__ w1,
                                    const float* __restrict__ b1,
                                    _Float16* __restrict__ out16) {
  int i = blockIdx.x * blockDim.x + threadIdx.x;
  if (i >= NN * HDIM) return;
  int n = i >> 7, j = i & 127;
  float s = b1[j];
#pragma unroll
  for (int k = 0; k < NIN; ++k) s += x[n * NIN + k] * w1[k * HDIM + j];
  out16[i] = (_Float16)fmaxf(s, 0.f);
}

// Decoder second layer: out[n] = b + sum_j t[n][j] * w[j]
__global__ void gnn_decoder2_kernel(const float* __restrict__ t,
                                    const float* __restrict__ w2,
                                    const float* __restrict__ b2,
                                    float* __restrict__ out) {
  int n = blockIdx.x * blockDim.x + threadIdx.x;
  if (n >= NN) return;
  float s = b2[0];
#pragma unroll 4
  for (int j = 0; j < HDIM; ++j) s += t[n * HDIM + j] * w2[j];
  out[n] = s;
}

// ---------------------------------------------------------------------------
// Generic 16-row x 128-col WMMA tile: out = (A16[16,128] @ B[128,128]) + bias.
// RELU / f16 mirror are compile-time flags (straight-line epilogue).
// 256 threads = 8 waves; wave w owns column tile w.
// ---------------------------------------------------------------------------
template <bool RELU, bool EMIT16>
__global__ __launch_bounds__(256)
void gnn_gemm128_tile_kernel(const _Float16* __restrict__ A16,
                             const _Float16* __restrict__ pB,
                             const float* __restrict__ bias,
                             float* __restrict__ out,
                             _Float16* __restrict__ out16) {
  __shared__ __align__(16) _Float16 sA[16 * 128];
  const int tid = threadIdx.x;
  const int r0  = blockIdx.x * 16;
  // pure f16 copy: 256 chunks of 8 halves, one per thread
  {
    int r = tid >> 4, s = tid & 15;
    *(v8h*)(sA + r * 128 + s * 8) =
        *(const v8h*)(A16 + (r0 + r) * HDIM + s * 8);
  }
  __syncthreads();
  const int wave = tid >> 5, lane = tid & 31;
  const int n = lane & 15, mbase = (lane & 16) ? 8 : 0;
  const int col = wave * 16 + n;
  v8f acc;
  { float bv = bias[col];
#pragma unroll
    for (int v = 0; v < 8; ++v) acc[v] = bv; }
#pragma unroll
  for (int kt = 0; kt < 4; ++kt) {
    v16h a = load_a_frag(sA, 128, kt, lane);
    v16h b = *(const v16h*)(pB + (kt * 8 + wave) * 512 + lane * 16);
    acc = __builtin_amdgcn_wmma_f32_16x16x32_f16(false, a, false, b,
                                                 (short)0, acc, false, false);
  }
#pragma unroll
  for (int v = 0; v < 8; ++v) {
    float f = acc[v];
    if (RELU) f = fmaxf(f, 0.f);
    out[(r0 + mbase + v) * HDIM + col] = f;
    if (EMIT16)
      out16[(r0 + mbase + v) * HDIM + col] = (_Float16)f;
  }
}

// ---------------------------------------------------------------------------
// Fused edge-message kernel, one block per 16 edges:
//   m_in = [h16[tgt] | h16[src] | edge_attr] (f16, K padded 259->288)
//   m = relu(m_in @ W1 + b1) @ W2 + b2
//   atomicAdd into agg[tgt]
// Gathers read the f16 shadow of h: halves the dominant HBM traffic.
// ---------------------------------------------------------------------------
__global__ __launch_bounds__(256)
void gnn_edge_msg_kernel(const _Float16* __restrict__ h16,
                         const int* __restrict__ src,
                         const int* __restrict__ tgt,
                         const float* __restrict__ eattr,
                         const _Float16* __restrict__ pW1,  // 9*8 frags
                         const _Float16* __restrict__ pW2,  // 4*8 frags
                         const float* __restrict__ b1,
                         const float* __restrict__ b2,
                         float* __restrict__ agg) {
  __shared__ __align__(16) _Float16 sA[16 * 288];
  __shared__ __align__(16) _Float16 sH1[16 * 128];
  __shared__ int sSrc[16], sTgt[16];
  const int tid = threadIdx.x;
  const int e0  = blockIdx.x * 16;
  if (tid < 16) { sSrc[tid] = src[e0 + tid]; sTgt[tid] = tgt[e0 + tid]; }
  __syncthreads();
  // gather h16: 512 chunks of 8 halves (16B), 2 per thread, fully vectorized
#pragma unroll
  for (int it = 0; it < 2; ++it) {
    int c = tid + it * 256;
    int r = c >> 5, s = c & 31;
    const _Float16* sp = (s < 16) ? (h16 + sTgt[r] * HDIM + s * 8)
                                  : (h16 + sSrc[r] * HDIM + (s - 16) * 8);
    *(v8h*)(sA + r * 288 + s * 8) = *(const v8h*)sp;
  }
  // edge_attr + zero pad: cols 256..287
#pragma unroll
  for (int it = 0; it < 2; ++it) {
    int c = tid + it * 256;
    int r = c >> 5, cc = c & 31;
    float v = (cc < EDIM) ? eattr[(e0 + r) * EDIM + cc] : 0.f;
    sA[r * 288 + 256 + cc] = (_Float16)v;
  }
  __syncthreads();

  const int wave = tid >> 5, lane = tid & 31;
  const int n = lane & 15, mbase = (lane & 16) ? 8 : 0;
  const int col = wave * 16 + n;

  v8f acc;
  { float bv = b1[col];
#pragma unroll
    for (int v = 0; v < 8; ++v) acc[v] = bv; }
#pragma unroll
  for (int kt = 0; kt < 9; ++kt) {
    v16h a = load_a_frag(sA, 288, kt, lane);
    v16h b = *(const v16h*)(pW1 + (kt * 8 + wave) * 512 + lane * 16);
    acc = __builtin_amdgcn_wmma_f32_16x16x32_f16(false, a, false, b,
                                                 (short)0, acc, false, false);
  }
#pragma unroll
  for (int v = 0; v < 8; ++v)
    sH1[(mbase + v) * 128 + col] = (_Float16)fmaxf(acc[v], 0.f);
  __syncthreads();

  v8f acc2;
  { float bv = b2[col];
#pragma unroll
    for (int v = 0; v < 8; ++v) acc2[v] = bv; }
#pragma unroll
  for (int kt = 0; kt < 4; ++kt) {
    v16h a = load_a_frag(sH1, 128, kt, lane);
    v16h b = *(const v16h*)(pW2 + (kt * 8 + wave) * 512 + lane * 16);
    acc2 = __builtin_amdgcn_wmma_f32_16x16x32_f16(false, a, false, b,
                                                  (short)0, acc2, false, false);
  }
#pragma unroll
  for (int v = 0; v < 8; ++v)
    atomicAdd(&agg[sTgt[mbase + v] * HDIM + col], acc2[v]);
}

// ---------------------------------------------------------------------------
// Fused node-update kernel, one block per 16 nodes:
//   u_in = [h16 | agg*inv]  (K=256)
//   u = relu(u_in @ W1 + b1) @ W2 + b2
//   h_out = layernorm(u + h) * g + b   (f32 master + f16 shadow)
// ---------------------------------------------------------------------------
__global__ __launch_bounds__(256)
void gnn_node_update_kernel(const float* __restrict__ h,
                            const _Float16* __restrict__ h16,
                            const float* __restrict__ agg,
                            const float* __restrict__ inv,
                            const _Float16* __restrict__ pW1,  // 8*8 frags
                            const _Float16* __restrict__ pW2,  // 4*8 frags
                            const float* __restrict__ b1,
                            const float* __restrict__ b2,
                            const float* __restrict__ lng,
                            const float* __restrict__ lnb,
                            float* __restrict__ h_out,
                            _Float16* __restrict__ h_out16) {
  __shared__ __align__(16) _Float16 sA[16 * 256];
  __shared__ __align__(16) _Float16 sH1[16 * 128];
  __shared__ __align__(16) float    sU[16 * 128];
  __shared__ float sMu[16], sRs[16];
  const int tid = threadIdx.x;
  const int n0  = blockIdx.x * 16;
  // cols 0..127: straight f16 copy of h16 (16B chunks, one per thread)
  {
    int r = tid >> 4, s = tid & 15;
    *(v8h*)(sA + r * 256 + s * 8) =
        *(const v8h*)(h16 + (n0 + r) * HDIM + s * 8);
  }
  // cols 128..255: agg * inv, f32 -> f16
  for (int idx = tid; idx < 16 * 128; idx += 256) {
    int r = idx >> 7, cc = idx & 127;
    sA[r * 256 + 128 + cc] = (_Float16)(agg[(n0 + r) * HDIM + cc] * inv[n0 + r]);
  }
  __syncthreads();

  const int wave = tid >> 5, lane = tid & 31;
  const int n = lane & 15, mbase = (lane & 16) ? 8 : 0;
  const int col = wave * 16 + n;

  v8f acc;
  { float bv = b1[col];
#pragma unroll
    for (int v = 0; v < 8; ++v) acc[v] = bv; }
#pragma unroll
  for (int kt = 0; kt < 8; ++kt) {
    v16h a = load_a_frag(sA, 256, kt, lane);
    v16h b = *(const v16h*)(pW1 + (kt * 8 + wave) * 512 + lane * 16);
    acc = __builtin_amdgcn_wmma_f32_16x16x32_f16(false, a, false, b,
                                                 (short)0, acc, false, false);
  }
#pragma unroll
  for (int v = 0; v < 8; ++v)
    sH1[(mbase + v) * 128 + col] = (_Float16)fmaxf(acc[v], 0.f);
  __syncthreads();

  v8f acc2;
  { float bv = b2[col];
#pragma unroll
    for (int v = 0; v < 8; ++v) acc2[v] = bv; }
#pragma unroll
  for (int kt = 0; kt < 4; ++kt) {
    v16h a = load_a_frag(sH1, 128, kt, lane);
    v16h b = *(const v16h*)(pW2 + (kt * 8 + wave) * 512 + lane * 16);
    acc2 = __builtin_amdgcn_wmma_f32_16x16x32_f16(false, a, false, b,
                                                  (short)0, acc2, false, false);
  }
  // residual into LDS (f32, from the f32 master copy of h)
#pragma unroll
  for (int v = 0; v < 8; ++v) {
    int m = mbase + v;
    sU[m * 128 + col] = acc2[v] + h[(n0 + m) * HDIM + col];
  }
  __syncthreads();
  // LayerNorm stats: 16 threads per row (one 16-lane group inside a wave),
  // each sums 8 elements, then shfl-xor tree reduce; single pass mean/E[x^2].
  {
    int r = tid >> 4, s = tid & 15;
    float s1 = 0.f, s2 = 0.f;
#pragma unroll
    for (int i = 0; i < 8; ++i) {
      float u = sU[r * 128 + s * 8 + i];
      s1 += u;
      s2 += u * u;
    }
#pragma unroll
    for (int off = 8; off > 0; off >>= 1) {
      s1 += __shfl_xor(s1, off, 32);
      s2 += __shfl_xor(s2, off, 32);
    }
    if (s == 0) {
      float mu  = s1 * (1.f / 128.f);
      float var = s2 * (1.f / 128.f) - mu * mu;
      sMu[r] = mu;
      sRs[r] = rsqrtf(var + LN_EPS);
    }
  }
  __syncthreads();
  for (int idx = tid; idx < 16 * 128; idx += 256) {
    int r = idx >> 7, c = idx & 127;
    float o = (sU[idx] - sMu[r]) * sRs[r] * lng[c] + lnb[c];
    h_out[(n0 + r) * HDIM + c] = o;
    h_out16[(n0 + r) * HDIM + c] = (_Float16)o;
  }
}

// ---------------------------------------------------------------------------
// Host-side launch sequence (graph-capture safe: kernels only).
// ---------------------------------------------------------------------------
extern "C" void kernel_launch(void* const* d_in, const int* in_sizes, int n_in,
                              void* d_out, int out_size, void* d_ws, size_t ws_size,
                              hipStream_t stream) {
  (void)in_sizes; (void)n_in; (void)out_size; (void)ws_size;
  const float* x      = (const float*)d_in[0];
  const int*   ei     = (const int*)  d_in[1];
  const float* eattr  = (const float*)d_in[2];
  const float* enc_w1 = (const float*)d_in[3];
  const float* enc_b1 = (const float*)d_in[4];
  const float* enc_w2 = (const float*)d_in[5];
  const float* enc_b2 = (const float*)d_in[6];
  const float* msg_w1 = (const float*)d_in[7];
  const float* msg_b1 = (const float*)d_in[8];
  const float* msg_w2 = (const float*)d_in[9];
  const float* msg_b2 = (const float*)d_in[10];
  const float* upd_w1 = (const float*)d_in[11];
  const float* upd_b1 = (const float*)d_in[12];
  const float* upd_w2 = (const float*)d_in[13];
  const float* upd_b2 = (const float*)d_in[14];
  const float* ln_g   = (const float*)d_in[15];
  const float* ln_b   = (const float*)d_in[16];
  const float* dec_w1 = (const float*)d_in[17];
  const float* dec_b1 = (const float*)d_in[18];
  const float* dec_w2 = (const float*)d_in[19];
  const float* dec_b2 = (const float*)d_in[20];

  const int* srcIdx = ei;        // edge_index[0]
  const int* tgtIdx = ei + NE;   // edge_index[1]

  // Carve workspace (256B aligned segments).
  char* p = (char*)d_ws;
  auto carve = [&](size_t bytes) -> void* {
    void* r = (void*)p;
    p += (bytes + 255) & ~(size_t)255;
    return r;
  };
  float*    counts = (float*)carve(NN * 4);
  float*    inv    = (float*)carve(NN * 4);
  float*    hA     = (float*)carve((size_t)NN * HDIM * 4);
  float*    hB     = (float*)carve((size_t)NN * HDIM * 4);
  float*    agg    = (float*)carve((size_t)NN * HDIM * 4);
  float*    tmp1   = (float*)carve((size_t)NN * HDIM * 4);
  _Float16* hA16   = (_Float16*)carve((size_t)NN * HDIM * 2);
  _Float16* hB16   = (_Float16*)carve((size_t)NN * HDIM * 2);
  _Float16* tmp16  = (_Float16*)carve((size_t)NN * HDIM * 2);
  _Float16* pEnc2  = (_Float16*)carve((size_t)128 * 128 * 2);
  _Float16* pMsgW1 = (_Float16*)carve((size_t)NL * 288 * 128 * 2);
  _Float16* pMsgW2 = (_Float16*)carve((size_t)NL * 128 * 128 * 2);
  _Float16* pUpdW1 = (_Float16*)carve((size_t)NL * 256 * 128 * 2);
  _Float16* pUpdW2 = (_Float16*)carve((size_t)NL * 128 * 128 * 2);
  _Float16* pDecW1 = (_Float16*)carve((size_t)128 * 128 * 2);

  // ---- pack all weights to f16 B-fragment layout ----
  gnn_pack_b_kernel<<<32, 32, 0, stream>>>(enc_w2, pEnc2, 128, 128);
  gnn_pack_b_kernel<<<32, 32, 0, stream>>>(dec_w1, pDecW1, 128, 128);
  for (int l = 0; l < NL; ++l) {
    gnn_pack_b_kernel<<<72, 32, 0, stream>>>(msg_w1 + (size_t)l * 259 * 128,
                                             pMsgW1 + (size_t)l * 288 * 128, 259, 128);
    gnn_pack_b_kernel<<<32, 32, 0, stream>>>(msg_w2 + (size_t)l * 128 * 128,
                                             pMsgW2 + (size_t)l * 128 * 128, 128, 128);
    gnn_pack_b_kernel<<<64, 32, 0, stream>>>(upd_w1 + (size_t)l * 256 * 128,
                                             pUpdW1 + (size_t)l * 256 * 128, 256, 128);
    gnn_pack_b_kernel<<<32, 32, 0, stream>>>(upd_w2 + (size_t)l * 128 * 128,
                                             pUpdW2 + (size_t)l * 128 * 128, 128, 128);
  }

  // ---- degree counts -> reciprocals ----
  gnn_zero_kernel<<<(NN + 255) / 256, 256, 0, stream>>>(counts, NN);
  gnn_count_kernel<<<(NE + 255) / 256, 256, 0, stream>>>(tgtIdx, counts);
  gnn_inv_kernel<<<(NN + 255) / 256, 256, 0, stream>>>(counts, inv);

  // ---- encoder ----
  gnn_encoder1_kernel<<<(NN * HDIM + 255) / 256, 256, 0, stream>>>(x, enc_w1,
                                                                   enc_b1, tmp16);
  gnn_gemm128_tile_kernel<false, true><<<NN / 16, 256, 0, stream>>>(
      tmp16, pEnc2, enc_b2, hA, hA16);

  // ---- message-passing layers ----
  float*    hcur   = hA;   float*    hnxt   = hB;
  _Float16* hcur16 = hA16; _Float16* hnxt16 = hB16;
  for (int l = 0; l < NL; ++l) {
    gnn_zero_kernel<<<(NN * HDIM + 255) / 256, 256, 0, stream>>>(agg, NN * HDIM);
    gnn_edge_msg_kernel<<<NE / 16, 256, 0, stream>>>(
        hcur16, srcIdx, tgtIdx, eattr,
        pMsgW1 + (size_t)l * 288 * 128, pMsgW2 + (size_t)l * 128 * 128,
        msg_b1 + l * HDIM, msg_b2 + l * HDIM, agg);
    gnn_node_update_kernel<<<NN / 16, 256, 0, stream>>>(
        hcur, hcur16, agg, inv,
        pUpdW1 + (size_t)l * 256 * 128, pUpdW2 + (size_t)l * 128 * 128,
        upd_b1 + l * HDIM, upd_b2 + l * HDIM,
        ln_g + l * HDIM, ln_b + l * HDIM, hnxt, hnxt16);
    float* t = hcur; hcur = hnxt; hnxt = t;
    _Float16* t16 = hcur16; hcur16 = hnxt16; hnxt16 = t16;
  }

  // ---- decoder ----
  gnn_gemm128_tile_kernel<true, false><<<NN / 16, 256, 0, stream>>>(
      hcur16, pDecW1, dec_b1, tmp1, nullptr);
  gnn_decoder2_kernel<<<(NN + 127) / 128, 128, 0, stream>>>(tmp1, dec_w2, dec_b2,
                                                            (float*)d_out);
}
